// MixtralMoE_50483045597490
// MI455X (gfx1250) — matrix operand order repaired and checked by reading, not compiled, not run on hardware
//
#include <hip/hip_runtime.h>

// ---------------- problem constants ----------------
#define NUM_EXPERTS 8
#define HIDDEN      2048
#define INTER       5632
#define NTOK        8192        // B*S = 4*2048

// ---------------- tiling ----------------
#define BM 128
#define BN 64
#define BK 32
#define LDSTR 40                // BK + 8 pad (bf16 elems) = 80B rows, 16B aligned

typedef __bf16        bf16x16 __attribute__((ext_vector_type(16)));
typedef __bf16        bf16x4  __attribute__((ext_vector_type(4)));
typedef float         f32x8   __attribute__((ext_vector_type(8)));
typedef float         f32x4   __attribute__((ext_vector_type(4)));
typedef unsigned int  u32x4   __attribute__((ext_vector_type(4)));
typedef unsigned int  u32x2   __attribute__((ext_vector_type(2)));

__device__ __forceinline__ unsigned short f2bf(float f) {
  union { __bf16 b; unsigned short u; } r;
  r.b = (__bf16)f;                        // native v_cvt (RNE)
  return r.u;
}

// pack 4 f32 -> 4 bf16 (backend: v_cvt_pk_bf16_f32 pairs)
__device__ __forceinline__ u32x2 pack4bf(f32x4 v) {
  union { bf16x4 b; u32x2 u; } r;
  r.b = __builtin_convertvector(v, bf16x4);
  return r.u;
}

// A fragment: 16 contiguous bf16 at p
__device__ __forceinline__ bf16x16 load_fragA(const unsigned short* p) {
  union { bf16x16 v; u32x4 q[2]; } u;
  u.q[0] = *(const u32x4*)(p);
  u.q[1] = *(const u32x4*)(p + 8);
  return u.v;
}
// B fragment: 16 contiguous bf16 at p
__device__ __forceinline__ bf16x16 load_fragB(const unsigned short* p) {
  union { bf16x16 v; u32x4 q[2]; } u;
  u.q[0] = *(const u32x4*)(p);
  u.q[1] = *(const u32x4*)(p + 8);
  return u.v;
}

// ---------------- router: logits -> softmax -> top2 -> gather lists ----------------
__global__ __launch_bounds__(256)
void moe_router(const float* __restrict__ x, const float* __restrict__ Wg,
                int* __restrict__ counts, int* __restrict__ rowlist,
                float* __restrict__ wtlist) {
  const int lane = threadIdx.x & 31;
  const int wave = threadIdx.x >> 5;
  const int t = blockIdx.x * 8 + wave;       // one wave32 per token
  const float* xt = x + (size_t)t * HIDDEN;

  float acc[NUM_EXPERTS];
#pragma unroll
  for (int e = 0; e < NUM_EXPERTS; ++e) acc[e] = 0.f;

  for (int k = lane; k < HIDDEN; k += 32) {
    float xv = xt[k];
#pragma unroll
    for (int e = 0; e < NUM_EXPERTS; ++e) acc[e] += xv * Wg[e * HIDDEN + k];
  }
#pragma unroll
  for (int off = 16; off > 0; off >>= 1)
#pragma unroll
    for (int e = 0; e < NUM_EXPERTS; ++e) acc[e] += __shfl_xor(acc[e], off, 32);

  if (lane == 0) {
    float mx = acc[0];
#pragma unroll
    for (int e = 1; e < NUM_EXPERTS; ++e) mx = fmaxf(mx, acc[e]);
    float p[NUM_EXPERTS];
#pragma unroll
    for (int e = 0; e < NUM_EXPERTS; ++e) p[e] = __expf(acc[e] - mx);
    // top-2 (softmax denominator cancels in the renormalization)
    int i0 = 0; float v0 = p[0];
#pragma unroll
    for (int e = 1; e < NUM_EXPERTS; ++e) if (p[e] > v0) { v0 = p[e]; i0 = e; }
    int i1 = -1; float v1 = -1.f;
#pragma unroll
    for (int e = 0; e < NUM_EXPERTS; ++e)
      if (e != i0 && p[e] > v1) { v1 = p[e]; i1 = e; }
    float inv = 1.f / (v0 + v1);
    int s0 = atomicAdd(&counts[i0], 1);
    rowlist[i0 * NTOK + s0] = t; wtlist[i0 * NTOK + s0] = v0 * inv;
    int s1 = atomicAdd(&counts[i1], 1);
    rowlist[i1 * NTOK + s1] = t; wtlist[i1 * NTOK + s1] = v1 * inv;
  }
}

__global__ void moe_scan(const int* __restrict__ counts, int* __restrict__ bases) {
  if (threadIdx.x == 0) {
    int s = 0;
#pragma unroll
    for (int e = 0; e < NUM_EXPERTS; ++e) { bases[e] = s; s += counts[e]; }
  }
}

// ---------------- GEMM1: h = silu(x@w1^T) * (x@w3^T), bf16 out ----------------
__global__ __launch_bounds__(256)
void moe_gemm1(const float* __restrict__ x, const float* __restrict__ w1,
               const float* __restrict__ w3, const int* __restrict__ counts,
               const int* __restrict__ bases, const int* __restrict__ rowlist,
               unsigned short* __restrict__ hbuf) {
  const int e = blockIdx.z;
  const int count = counts[e];
  const int m0 = blockIdx.y * BM;
  if (m0 >= count) return;
  const int n0 = blockIdx.x * BN;

  __shared__ unsigned short As[BM * LDSTR];
  __shared__ unsigned short Bs1[BN * LDSTR];
  __shared__ unsigned short Bs3[BN * LDSTR];

  const int tid  = threadIdx.x;
  const int lane = tid & 31;
  const int wid  = tid >> 5;
  const int wm   = (wid & 3) * 32;
  const int wn   = (wid >> 2) * 32;
  const int frow = lane & 15;
  const int kA   = (lane >> 4) << 4;     // A: K base 0/16
  const int kB   = (lane >> 4) << 4;     // B: K base 0/16

  const int lr = tid >> 3;               // 0..31 (loader row)
  const int lc = (tid & 7) << 2;         // 0..28 (loader col, 4 f32)

  // Clamp out-of-range tile rows to a valid row (no divergent loads; the
  // epilogue masks those M rows anyway).
  const float* arow[4];
#pragma unroll
  for (int p = 0; p < 4; ++p) {
    int m = m0 + p * 32 + lr;
    if (m >= count) m = count - 1;
    arow[p] = x + (size_t)rowlist[e * NTOK + m] * HIDDEN + lc;
  }
  const float* w1e = w1 + (size_t)e * INTER * HIDDEN;
  const float* w3e = w3 + (size_t)e * INTER * HIDDEN;

  const f32x8 z8 = {0.f,0.f,0.f,0.f,0.f,0.f,0.f,0.f};
  f32x8 acc1[2][2], acc3[2][2];
#pragma unroll
  for (int i = 0; i < 2; ++i)
#pragma unroll
    for (int j = 0; j < 2; ++j) { acc1[i][j] = z8; acc3[i][j] = z8; }

  for (int k0 = 0; k0 < HIDDEN; k0 += BK) {
#pragma unroll
    for (int p = 0; p < 4; ++p) {                       // A: 128x32 f32 -> bf16
      int r = p * 32 + lr;
      f32x4 v = *(const f32x4*)(arow[p] + k0);
      *(u32x2*)&As[r * LDSTR + lc] = pack4bf(v);
    }
#pragma unroll
    for (int p = 0; p < 2; ++p) {                       // B1/B3: 64x32 f32 -> bf16
      int r = p * 32 + lr;
      f32x4 v1 = *(const f32x4*)(w1e + (size_t)(n0 + r) * HIDDEN + k0 + lc);
      f32x4 v3 = *(const f32x4*)(w3e + (size_t)(n0 + r) * HIDDEN + k0 + lc);
      *(u32x2*)&Bs1[r * LDSTR + lc] = pack4bf(v1);
      *(u32x2*)&Bs3[r * LDSTR + lc] = pack4bf(v3);
    }
    if (k0 + BK < HIDDEN) {                             // prefetch next weight tiles
      __builtin_prefetch(w1e + (size_t)(n0 + lr) * HIDDEN + k0 + BK + lc, 0, 1);
      __builtin_prefetch(w3e + (size_t)(n0 + lr) * HIDDEN + k0 + BK + lc, 0, 1);
    }
    __syncthreads();

    bf16x16 a[2], b1[2], b3[2];
#pragma unroll
    for (int t = 0; t < 2; ++t) {
      a[t]  = load_fragA(&As [(wm + t * 16 + frow) * LDSTR + kA]);
      b1[t] = load_fragB(&Bs1[(wn + t * 16 + frow) * LDSTR + kB]);
      b3[t] = load_fragB(&Bs3[(wn + t * 16 + frow) * LDSTR + kB]);
    }
#pragma unroll
    for (int tm = 0; tm < 2; ++tm)
#pragma unroll
      for (int tn = 0; tn < 2; ++tn) {
        acc1[tm][tn] = __builtin_amdgcn_wmma_f32_16x16x32_bf16(
            false, a[tm], false, b1[tn], (short)0, acc1[tm][tn], false, false);
        acc3[tm][tn] = __builtin_amdgcn_wmma_f32_16x16x32_bf16(
            false, a[tm], false, b3[tn], (short)0, acc3[tm][tn], false, false);
      }
    __syncthreads();
  }

  const int hbase = bases[e];
  const int mlane = (lane >> 4) << 3;
  const int nlane = lane & 15;
#pragma unroll
  for (int tm = 0; tm < 2; ++tm)
#pragma unroll
    for (int j = 0; j < 8; ++j) {
      int m = m0 + wm + tm * 16 + mlane + j;
      if (m < count) {
        size_t rb = (size_t)(hbase + m) * INTER + n0 + wn;
#pragma unroll
        for (int tn = 0; tn < 2; ++tn) {
          float s1 = acc1[tm][tn][j], s3 = acc3[tm][tn][j];
          float h = (s1 / (1.f + __expf(-s1))) * s3;    // silu(s1)*s3
          hbuf[rb + tn * 16 + nlane] = f2bf(h);
        }
      }
    }
}

// ---------------- GEMM2: out += wt * (h @ w2^T), atomic scatter ----------------
__global__ __launch_bounds__(256)
void moe_gemm2(const unsigned short* __restrict__ hbuf, const float* __restrict__ w2,
               const int* __restrict__ counts, const int* __restrict__ bases,
               const int* __restrict__ rowlist, const float* __restrict__ wtlist,
               float* __restrict__ out) {
  const int e = blockIdx.z;
  const int count = counts[e];
  const int m0 = blockIdx.y * BM;
  if (m0 >= count) return;
  const int n0 = blockIdx.x * BN;          // over HIDDEN
  const int hbase = bases[e];

  __shared__ unsigned short As[BM * LDSTR];
  __shared__ unsigned short Bs[BN * LDSTR];

  const int tid  = threadIdx.x;
  const int lane = tid & 31;
  const int wid  = tid >> 5;
  const int wm   = (wid & 3) * 32;
  const int wn   = (wid >> 2) * 32;
  const int frow = lane & 15;
  const int kA   = (lane >> 4) << 4;
  const int kB   = (lane >> 4) << 4;

  const int ar = tid >> 2;                 // A loader: 0..63 rows, 8 bf16 each
  const int ac = (tid & 3) << 3;
  const int br = tid >> 3;                 // B loader: 0..31 rows, 4 f32 each
  const int bc = (tid & 7) << 2;

  // clamped A row pointers (already bf16 in hbuf)
  const unsigned short* ah[2];
#pragma unroll
  for (int p = 0; p < 2; ++p) {
    int m = m0 + p * 64 + ar;
    if (m >= count) m = count - 1;
    ah[p] = hbuf + (size_t)(hbase + m) * INTER + ac;
  }
  const float* w2e = w2 + (size_t)e * HIDDEN * INTER;

  const f32x8 z8 = {0.f,0.f,0.f,0.f,0.f,0.f,0.f,0.f};
  f32x8 acc[2][2];
#pragma unroll
  for (int i = 0; i < 2; ++i)
#pragma unroll
    for (int j = 0; j < 2; ++j) acc[i][j] = z8;

  for (int k0 = 0; k0 < INTER; k0 += BK) {
#pragma unroll
    for (int p = 0; p < 2; ++p) {                       // A: 128x32 bf16 direct
      int r = p * 64 + ar;
      *(u32x4*)&As[r * LDSTR + ac] = *(const u32x4*)(ah[p] + k0);
    }
#pragma unroll
    for (int p = 0; p < 2; ++p) {                       // B: 64x32 f32 -> bf16
      int r = p * 32 + br;
      f32x4 v = *(const f32x4*)(w2e + (size_t)(n0 + r) * INTER + k0 + bc);
      *(u32x2*)&Bs[r * LDSTR + bc] = pack4bf(v);
    }
    if (k0 + BK < INTER)                                // prefetch next weight tile
      __builtin_prefetch(w2e + (size_t)(n0 + br) * INTER + k0 + BK + bc, 0, 1);
    __syncthreads();

    bf16x16 a[2], b[2];
#pragma unroll
    for (int t = 0; t < 2; ++t) {
      a[t] = load_fragA(&As[(wm + t * 16 + frow) * LDSTR + kA]);
      b[t] = load_fragB(&Bs[(wn + t * 16 + frow) * LDSTR + kB]);
    }
#pragma unroll
    for (int tm = 0; tm < 2; ++tm)
#pragma unroll
      for (int tn = 0; tn < 2; ++tn)
        acc[tm][tn] = __builtin_amdgcn_wmma_f32_16x16x32_bf16(
            false, a[tm], false, b[tn], (short)0, acc[tm][tn], false, false);
    __syncthreads();
  }

  const int mlane = (lane >> 4) << 3;
  const int nlane = lane & 15;
#pragma unroll
  for (int tm = 0; tm < 2; ++tm)
#pragma unroll
    for (int j = 0; j < 8; ++j) {
      int m = m0 + wm + tm * 16 + mlane + j;
      if (m < count) {
        int tok = rowlist[e * NTOK + m];
        float wt = wtlist[e * NTOK + m];
#pragma unroll
        for (int tn = 0; tn < 2; ++tn)
          atomicAdd(&out[(size_t)tok * HIDDEN + n0 + wn + tn * 16 + nlane],
                    wt * acc[tm][tn][j]);
      }
    }
}

// ---------------- launcher ----------------
extern "C" void kernel_launch(void* const* d_in, const int* in_sizes, int n_in,
                              void* d_out, int out_size, void* d_ws, size_t ws_size,
                              hipStream_t stream) {
  const float* x  = (const float*)d_in[0];
  const float* Wg = (const float*)d_in[1];
  const float* w1 = (const float*)d_in[2];
  const float* w3 = (const float*)d_in[3];
  const float* w2 = (const float*)d_in[4];
  float* out = (float*)d_out;

  // workspace layout
  char* ws = (char*)d_ws;
  const size_t LIST_BYTES = (size_t)NUM_EXPERTS * NTOK * 4;     // 256 KB each
  int*   counts  = (int*)ws;                                    // [8]
  int*   bases   = (int*)(ws + 32);                             // [8]
  int*   rowlist = (int*)(ws + 64);
  float* wtlist  = (float*)(ws + 64 + LIST_BYTES);
  unsigned short* hbuf = (unsigned short*)(ws + 64 + 2 * LIST_BYTES);
  const size_t need = 64 + 2 * LIST_BYTES + (size_t)2 * NTOK * INTER * 2; // ~176.6 MB

  hipMemsetAsync(d_out, 0, (size_t)out_size * sizeof(float), stream);
  if (ws_size < need) return;                 // deterministic (zero) fallback
  hipMemsetAsync(ws, 0, 64, stream);

  moe_router<<<NTOK / 8, 256, 0, stream>>>(x, Wg, counts, rowlist, wtlist);
  moe_scan<<<1, 32, 0, stream>>>(counts, bases);
  moe_gemm1<<<dim3(INTER / BN, NTOK / BM, NUM_EXPERTS), 256, 0, stream>>>(
      x, w1, w3, counts, bases, rowlist, hbuf);
  moe_gemm2<<<dim3(HIDDEN / BN, NTOK / BM, NUM_EXPERTS), 256, 0, stream>>>(
      hbuf, w2, counts, bases, rowlist, wtlist, out);
}
